// GIN_54296976556546
// MI455X (gfx1250) — compile-verified, matrix operand order checked by reading, not run on hardware
//
#include <hip/hip_runtime.h>
#include <hip/hip_bf16.h>

#define GIN_N 50000
#define GIN_E 800000

typedef __attribute__((ext_vector_type(16))) _Float16 v16h;
typedef __attribute__((ext_vector_type(8)))  float    v8f;

// -------------------------------------------------------------------------
// Edge scatter: z[dst] += h[src], float4-vectorized; atomics land in L2
// (all node-feature matrices are <= 51MB, L2 is 192MB).
// -------------------------------------------------------------------------
__global__ void __launch_bounds__(256)
scatter_add4(float* __restrict__ z, const float* __restrict__ h,
             const int* __restrict__ src, const int* __restrict__ dst,
             int nedges, int D)
{
    int tid    = blockIdx.x * blockDim.x + threadIdx.x;
    int chunks = D >> 2;
    int total  = nedges * chunks;          // <= 51.2M
    if (tid >= total) return;
    int e = tid / chunks;
    int c = (tid - e * chunks) << 2;
    int s = src[e];
    int d = dst[e];
    const float4 v = *(const float4*)(h + (size_t)s * D + c);
    float* p = z + (size_t)d * D + c;
    atomicAdd(p + 0, v.x);
    atomicAdd(p + 1, v.y);
    atomicAdd(p + 2, v.z);
    atomicAdd(p + 3, v.w);
}

// -------------------------------------------------------------------------
// Fragment packers: write A / B in CDNA5 WMMA per-lane order so the GEMM
// inner loop does pure 32B-contiguous vector loads (b128 pairs) + WMMA.
// ISA 16-bit A layout: lanes 0-15 -> M=0..15 with K pairs {2v,2v+1} (v<4)
// and {16+2(v-4), ...}; lanes 16-31 same shifted by K+=8. B is symmetric
// (lane = column, same K pattern). One v16h (32B) per lane per K-step.
// -------------------------------------------------------------------------
__global__ void __launch_bounds__(256)
pack_a_f16(const float* __restrict__ A, v16h* __restrict__ P,
           int K, int ktiles, int total)
{
    int tid = blockIdx.x * blockDim.x + threadIdx.x;
    if (tid >= total) return;
    const int lane = tid & 31;
    const int t    = tid >> 5;
    const int kt   = t % ktiles;
    const int mt   = t / ktiles;
    const float* row = A + (size_t)(mt * 16 + (lane & 15)) * K
                         + kt * 32 + ((lane >> 4) << 3);
    v16h o;
#pragma unroll
    for (int v = 0; v < 8; ++v) {
        const int k = (v < 4) ? (2 * v) : (16 + 2 * (v - 4));
        o[2 * v]     = (_Float16)row[k];
        o[2 * v + 1] = (_Float16)row[k + 1];
    }
    P[tid] = o;
}

__global__ void __launch_bounds__(256)
pack_b_f16(const float* __restrict__ W, v16h* __restrict__ P,
           int Nout, int ntiles, int total)
{
    int tid = blockIdx.x * blockDim.x + threadIdx.x;
    if (tid >= total) return;
    const int lane = tid & 31;
    const int t    = tid >> 5;
    const int nt   = t % ntiles;
    const int kt   = t / ntiles;
    const float* col = W + (size_t)(kt * 32 + ((lane >> 4) << 3)) * Nout
                         + nt * 16 + (lane & 15);
    v16h o;
#pragma unroll
    for (int v = 0; v < 8; ++v) {
        const int k = (v < 4) ? (2 * v) : (16 + 2 * (v - 4));
        o[2 * v]     = (_Float16)col[(size_t)k * Nout];
        o[2 * v + 1] = (_Float16)col[(size_t)(k + 1) * Nout];
    }
    P[tid] = o;
}

// -------------------------------------------------------------------------
// Packed WMMA GEMM with 2x2 register blocking: each wave computes a 32x32
// output region (4 f32 accumulators). Per K-step: 4 fragment loads (8x
// b128) feed 4x v_wmma_f32_16x16x32_f16 -> 2 loads per WMMA.
//   mode 0: relu(x + bias)
//   mode 1: relu((x + bias) * sc + sh)   (folded BatchNorm)
// mtiles is odd (3125): the last super-row aliases its second A-row to the
// first (wave-uniform select, EXEC stays all-1s) and skips its stores.
// -------------------------------------------------------------------------
__global__ void __launch_bounds__(256)
gin_wmma_gemm_2x2(const v16h* __restrict__ Ap, const v16h* __restrict__ Bp,
                  const float* __restrict__ bias, float* __restrict__ Cout,
                  int M, int ktiles, int Nout, int mode,
                  const float* __restrict__ bng, const float* __restrict__ bnb,
                  const float* __restrict__ bnm, const float* __restrict__ bnv)
{
    const int lane    = threadIdx.x & 31;
    const int wave    = threadIdx.x >> 5;
    const int ntiles  = Nout >> 4;
    const int mtiles  = M >> 4;
    const int n2      = ntiles >> 1;           // always even ntiles
    const int m2      = (mtiles + 1) >> 1;
    const int stile   = blockIdx.x * 8 + wave;
    if (stile >= m2 * n2) return;              // wave-uniform exit

    const int sm  = stile / n2;
    const int sn  = stile % n2;
    const int mt0 = sm * 2;
    const int mt1 = mt0 + 1;
    const int nt0 = sn * 2;
    const int nt1 = nt0 + 1;
    const bool m1ok = (mt1 < mtiles);          // wave-uniform

    const v16h* ap0 = Ap + ((size_t)mt0 * ktiles) * 32 + lane;
    const v16h* ap1 = Ap + ((size_t)(m1ok ? mt1 : mt0) * ktiles) * 32 + lane;
    const v16h* bp0 = Bp + (size_t)nt0 * 32 + lane;
    const v16h* bp1 = Bp + (size_t)nt1 * 32 + lane;
    const size_t bstride = (size_t)ntiles * 32;

    v8f acc00 = {}, acc01 = {}, acc10 = {}, acc11 = {};
#pragma unroll 2
    for (int kt = 0; kt < ktiles; ++kt) {
        const v16h a0 = ap0[(size_t)kt * 32];
        const v16h a1 = ap1[(size_t)kt * 32];
        const v16h b0 = bp0[(size_t)kt * bstride];
        const v16h b1 = bp1[(size_t)kt * bstride];
        acc00 = __builtin_amdgcn_wmma_f32_16x16x32_f16(
                    false, a0, false, b0, (short)0, acc00, false, false);
        acc01 = __builtin_amdgcn_wmma_f32_16x16x32_f16(
                    false, a0, false, b1, (short)0, acc01, false, false);
        acc10 = __builtin_amdgcn_wmma_f32_16x16x32_f16(
                    false, a1, false, b0, (short)0, acc10, false, false);
        acc11 = __builtin_amdgcn_wmma_f32_16x16x32_f16(
                    false, a1, false, b1, (short)0, acc11, false, false);
    }

    const int l15  = lane & 15;
    const int hi   = lane >> 4;
    const int col0 = (nt0 << 4) + l15;
    const int col1 = (nt1 << 4) + l15;

    float bi0 = bias[col0], bi1 = bias[col1];
    float sc0 = 1.0f, sh0 = 0.0f, sc1 = 1.0f, sh1 = 0.0f;
    if (mode == 1) {
        const float rs0 = __frsqrt_rn(bnv[col0] + 1e-5f);
        const float rs1 = __frsqrt_rn(bnv[col1] + 1e-5f);
        sc0 = bng[col0] * rs0;  sh0 = bnb[col0] - bnm[col0] * sc0;
        sc1 = bng[col1] * rs1;  sh1 = bnb[col1] - bnm[col1] * sc1;
    }

    const int rowb0 = (mt0 << 4) + (hi << 3);
#pragma unroll
    for (int v = 0; v < 8; ++v) {
        float x0 = acc00[v] + bi0;
        float x1 = acc01[v] + bi1;
        if (mode == 1) { x0 = x0 * sc0 + sh0; x1 = x1 * sc1 + sh1; }
        x0 = x0 > 0.0f ? x0 : 0.0f;
        x1 = x1 > 0.0f ? x1 : 0.0f;
        float* r = Cout + (size_t)(rowb0 + v) * Nout;
        r[col0] = x0;
        r[col1] = x1;
    }
    if (m1ok) {                                 // wave-uniform branch
        const int rowb1 = (mt1 << 4) + (hi << 3);
#pragma unroll
        for (int v = 0; v < 8; ++v) {
            float x0 = acc10[v] + bi0;
            float x1 = acc11[v] + bi1;
            if (mode == 1) { x0 = x0 * sc0 + sh0; x1 = x1 * sc1 + sh1; }
            x0 = x0 > 0.0f ? x0 : 0.0f;
            x1 = x1 > 0.0f ? x1 : 0.0f;
            float* r = Cout + (size_t)(rowb1 + v) * Nout;
            r[col0] = x0;
            r[col1] = x1;
        }
    }
}

// -------------------------------------------------------------------------
// Classifier: out[Nx2] = h[Nx64] @ wc[64x2] + bc   (too skinny for WMMA)
// -------------------------------------------------------------------------
__global__ void __launch_bounds__(256)
final_linear(const float* __restrict__ h, const float* __restrict__ wc,
             const float* __restrict__ bc, float* __restrict__ out, int M)
{
    int i = blockIdx.x * blockDim.x + threadIdx.x;
    if (i >= M) return;
    const float* r = h + (size_t)i * 64;
    float a0 = bc[0], a1 = bc[1];
#pragma unroll
    for (int k = 0; k < 64; ++k) {
        const float x = r[k];
        a0 += x * wc[k * 2 + 0];
        a1 += x * wc[k * 2 + 1];
    }
    out[i * 2 + 0] = a0;
    out[i * 2 + 1] = a1;
}

// -------------------------------------------------------------------------
// Host-side helpers
// -------------------------------------------------------------------------
static void run_gemm(const float* A, const float* W, const float* bias,
                     float* Cout, int K, int Nout, int mode,
                     const float* g, const float* be,
                     const float* m, const float* v,
                     v16h* apack, v16h* bpack, hipStream_t stream)
{
    const int ktiles = K >> 5;
    const int ntiles = Nout >> 4;
    const int mtiles = GIN_N >> 4;
    {
        int total = mtiles * ktiles * 32;
        pack_a_f16<<<(total + 255) / 256, 256, 0, stream>>>(A, apack, K, ktiles, total);
    }
    {
        int total = ktiles * ntiles * 32;
        pack_b_f16<<<(total + 255) / 256, 256, 0, stream>>>(W, bpack, Nout, ntiles, total);
    }
    {
        int stiles = ((mtiles + 1) / 2) * (ntiles / 2);
        gin_wmma_gemm_2x2<<<(stiles + 7) / 8, 256, 0, stream>>>(
            apack, bpack, bias, Cout, GIN_N, ktiles, Nout, mode, g, be, m, v);
    }
}

static void run_gin_layer(const float* h_in, int Din, int Dout,
                          const float* wa, const float* ba,
                          const float* wb, const float* bb,
                          const float* g, const float* be,
                          const float* m, const float* v,
                          const int* src, const int* dst,
                          float* zbuf, float* tbuf, float* hout,
                          v16h* apack, v16h* bpack, hipStream_t stream)
{
    // z = h  (GIN self term, eps=0)
    hipMemcpyAsync(zbuf, h_in, (size_t)GIN_N * Din * sizeof(float),
                   hipMemcpyDeviceToDevice, stream);
    // z += sum of neighbors
    {
        int total = GIN_E * (Din >> 2);
        scatter_add4<<<(total + 255) / 256, 256, 0, stream>>>(
            zbuf, h_in, src, dst, GIN_E, Din);
    }
    // t = relu(z @ wa + ba)
    run_gemm(zbuf, wa, ba, tbuf, Din, Dout, 0,
             nullptr, nullptr, nullptr, nullptr, apack, bpack, stream);
    // h_out = relu(BN(t @ wb + bb))
    run_gemm(tbuf, wb, bb, hout, Dout, Dout, 1, g, be, m, v,
             apack, bpack, stream);
}

extern "C" void kernel_launch(void* const* d_in, const int* in_sizes, int n_in,
                              void* d_out, int out_size, void* d_ws, size_t ws_size,
                              hipStream_t stream)
{
    (void)in_sizes; (void)n_in; (void)out_size; (void)ws_size;

    const float* x   = (const float*)d_in[0];
    const int*   ei  = (const int*)d_in[1];
    const int*   src = ei;            // edge_index row 0
    const int*   dst = ei + GIN_E;    // edge_index row 1

    const float* P[32];
    for (int i = 0; i < 32; ++i) P[i] = (const float*)d_in[2 + i];
    const float* wc = (const float*)d_in[34];
    const float* bc = (const float*)d_in[35];

    // workspace layout (all 32B-aligned): 3x Nx256 f32 + f16 pack buffers
    float* bufA = (float*)d_ws;                          // h
    float* bufB = bufA + (size_t)GIN_N * 256;            // z
    float* bufC = bufB + (size_t)GIN_N * 256;            // t
    v16h*  apack = (v16h*)(bufC + (size_t)GIN_N * 256);  // 25.6 MB max
    v16h*  bpack = apack + (size_t)(GIN_N / 16) * 8 * 32;

    const int dims_in [4] = {256, 128, 256, 128};
    const int dims_out[4] = {128, 256, 128,  64};

    const float* h = x;
    for (int L = 0; L < 4; ++L) {
        const float** p = &P[8 * L];
        run_gin_layer(h, dims_in[L], dims_out[L],
                      p[0], p[1], p[2], p[3],        // wa ba wb bb
                      p[4], p[5], p[6], p[7],        // bn g/b/m/v
                      src, dst, bufB, bufC, bufA, apack, bpack, stream);
        h = bufA;
    }

    final_linear<<<(GIN_N + 255) / 256, 256, 0, stream>>>(
        h, wc, bc, (float*)d_out, GIN_N);
}
